// KNNAttention_55379308314874
// MI455X (gfx1250) — compile-verified, matrix operand order checked by reading, not run on hardware
//
#include <hip/hip_runtime.h>
#include <hip/hip_bf16.h>

// Problem constants (match reference)
#define B_ 2
#define T_ 2048
#define D_ 1024
#define H_ 16
#define HD_ 64
#define K_ 3
#define M_ 16384
#define SCALE_F 4096.0f   // D * sqrt(H)
#define NEG_LARGE -3.0e38f

typedef __attribute__((ext_vector_type(16))) _Float16 v16h;
typedef __attribute__((ext_vector_type(8)))  _Float16 v8h;
typedef __attribute__((ext_vector_type(8)))  float    v8f;
typedef __attribute__((ext_vector_type(4)))  int      v4i;

// ---------------------------------------------------------------------------
// Optional gfx1250 async global->LDS copy (ASYNCcnt-tracked). Guarded so the
// file compiles on toolchains without the builtin (and on the host pass).
// Probe feedback: params are `v4i AS1*` / `v4i AS3*` (non-const).
// ---------------------------------------------------------------------------
#if __has_builtin(__builtin_amdgcn_global_load_async_to_lds_b128)
#define HAVE_ASYNC_COPY 1
#else
#define HAVE_ASYNC_COPY 0
#endif

__device__ __forceinline__ void copy16_g2l(_Float16* lp, const _Float16* gp) {
#if HAVE_ASYNC_COPY
  __builtin_amdgcn_global_load_async_to_lds_b128(
      (__attribute__((address_space(1))) v4i*)gp,
      (__attribute__((address_space(3))) v4i*)lp, 0, 0);
#else
  *(v8h*)lp = *(const v8h*)gp;
#endif
}

__device__ __forceinline__ void async_wait_all() {
#if HAVE_ASYNC_COPY
#if __has_builtin(__builtin_amdgcn_s_wait_asynccnt)
  __builtin_amdgcn_s_wait_asynccnt(0);
#else
  asm volatile("s_wait_asynccnt 0x0" ::: "memory");
#endif
#endif
}

// ---------------------------------------------------------------------------
// Load a 16x32 (f16) A- or B-fragment row for CDNA5 WMMA from row-major data.
// Caller passes p = base_of_row + 8*half (half = lane>>4). Element i of the
// v16h maps to K = (i&7) + 8*half + 16*(i>>3): two 16-byte chunks at
// K-offsets {8*half, 16+8*half}.
// ---------------------------------------------------------------------------
__device__ __forceinline__ v16h ldfrag(const _Float16* p) {
  v8h lo = *(const v8h*)(p);
  v8h hi = *(const v8h*)(p + 16);
  v16h o;
#pragma unroll
  for (int i = 0; i < 8; ++i) { o[i] = lo[i]; o[i + 8] = hi[i]; }
  return o;
}

__device__ __forceinline__ v8f wmma_f16(v16h a, v16h b, v8f c) {
  return __builtin_amdgcn_wmma_f32_16x16x32_f16(false, a, false, b, (short)0, c,
                                                false, false);
}

// ---------------------------------------------------------------------------
// f32 -> f16 conversion (x and weight matrices)
// ---------------------------------------------------------------------------
__global__ __launch_bounds__(256) void f32_to_f16_kernel(
    const float* __restrict__ in, _Float16* __restrict__ out, size_t n) {
  size_t i = (size_t)blockIdx.x * blockDim.x + threadIdx.x;
  size_t stride = (size_t)gridDim.x * blockDim.x;
  for (; i < n; i += stride) out[i] = (_Float16)in[i];
}

// ---------------------------------------------------------------------------
// GEMM (NT): C[m,n] = sum_k A[m,k] * W[n,k] + bias[n]
// Block = 128 threads (4 waves). Block tile 64x64; wave tile 16x64.
// The 64x32 W tile is staged in LDS (double-buffered, async DMA when
// available) and shared by all 4 waves. A rows are read directly (per-wave).
// ---------------------------------------------------------------------------
#define WT_STRIDE 48  // f16 elements per LDS row (96B: 16B-aligned, bank-spread)

__device__ __forceinline__ void stage_wtile(const _Float16* __restrict__ W,
                                            int n0, int kk, int Kd,
                                            _Float16* lds, int tid) {
  // 64 rows x 32 f16 = 256 16B chunks; 128 threads x 2 chunks.
#pragma unroll
  for (int i = 0; i < 2; ++i) {
    const int c = tid * 2 + i;
    const int row = c >> 2;   // 0..63
    const int part = c & 3;   // 16B chunk within the 32-f16 K-slice
    copy16_g2l(lds + row * WT_STRIDE + part * 8,
               W + (size_t)(n0 + row) * Kd + kk + part * 8);
  }
}

template <bool OUT16>
__global__ __launch_bounds__(128) void gemm_nt_kernel(
    const _Float16* __restrict__ A, const _Float16* __restrict__ W,
    const float* __restrict__ bias, float* __restrict__ Cf,
    _Float16* __restrict__ Ch, int Mr, int N, int Kd) {
  __shared__ _Float16 wt[2][64 * WT_STRIDE];
  const int tid = threadIdx.x;
  const int lane = tid & 31;
  const int wave = tid >> 5;
  const int r = lane & 15;
  const int half = lane >> 4;
  const int m0 = blockIdx.y * 64 + wave * 16;
  const int n0 = blockIdx.x * 64;

  v8f acc[4] = {};
  const _Float16* arow = A + (size_t)(m0 + r) * Kd + half * 8;

  stage_wtile(W, n0, 0, Kd, &wt[0][0], tid);
  const int nk = Kd >> 5;
  for (int i = 0; i < nk; ++i) {
    async_wait_all();        // own tile-i chunks landed in LDS
    __syncthreads();         // everyone's landed; prior reads of next buf done
    if (i + 1 < nk)          // DMA tile i+1 overlaps tile-i compute
      stage_wtile(W, n0, (i + 1) << 5, Kd, &wt[(i + 1) & 1][0], tid);
    __builtin_prefetch(arow + ((i + 2) << 5), 0, 3);  // global_prefetch_b8
    v16h a = ldfrag(arow + (i << 5));
    const _Float16* wl = &wt[i & 1][0];
#pragma unroll
    for (int t = 0; t < 4; ++t) {
      v16h b = ldfrag(wl + (t * 16 + r) * WT_STRIDE + half * 8);
      acc[t] = wmma_f16(a, b, acc[t]);
    }
  }

  // C/D layout: VGPR g -> row m0 + 8*half + g ; lane -> col n0 + 16*t + r
#pragma unroll
  for (int t = 0; t < 4; ++t) {
    const int col = n0 + t * 16 + r;
    const float bv = bias ? bias[col] : 0.0f;
#pragma unroll
    for (int g = 0; g < 8; ++g) {
      const int row = m0 + half * 8 + g;
      const float val = acc[t][g] + bv;
      if (OUT16)
        Ch[(size_t)row * N + col] = (_Float16)val;
      else
        Cf[(size_t)row * N + col] = val;
    }
  }
}

// ---------------------------------------------------------------------------
// In-place L2 row normalization (q, k), one block per row of D elements.
// ---------------------------------------------------------------------------
__global__ __launch_bounds__(256) void normalize_rows_kernel(_Float16* x) {
  _Float16* p = x + (size_t)blockIdx.x * D_;
  float ssq = 0.0f;
  for (int i = threadIdx.x; i < D_; i += 256) {
    float v = (float)p[i];
    ssq += v * v;
  }
#pragma unroll
  for (int m = 16; m; m >>= 1) ssq += __shfl_xor(ssq, m, 32);
  __shared__ float red[8];
  if ((threadIdx.x & 31) == 0) red[threadIdx.x >> 5] = ssq;
  __syncthreads();
  float tot = 0.0f;
#pragma unroll
  for (int w = 0; w < 8; ++w) tot += red[w];
  const float inv = 1.0f / fmaxf(sqrtf(tot), 1e-12f);
  for (int i = threadIdx.x; i < D_; i += 256) p[i] = (_Float16)((float)p[i] * inv);
}

// ---------------------------------------------------------------------------
// Transpose V: [b,t,D] -> [b,h,hd,t] so the token axis is contiguous for the
// P@V B-matrix fragments.
// ---------------------------------------------------------------------------
__global__ __launch_bounds__(256) void transpose_v_kernel(
    const _Float16* __restrict__ v, _Float16* __restrict__ vt) {
  const int bt = blockIdx.x;
  const int b = bt / T_;
  const int t = bt % T_;
  const _Float16* src = v + (size_t)bt * D_;
  for (int d = threadIdx.x; d < D_; d += 256) {
    const int h = d >> 6, hd = d & 63;
    vt[(((size_t)b * H_ + h) * HD_ + hd) * T_ + t] = src[d];
  }
}

// ---------------------------------------------------------------------------
// One 32-key flash-attention step for a 16-row query tile of one wave.
// MASKED instantiation only for the single diagonal block.
// Row sums of P come from an extra WMMA against an all-ones B fragment
// (lands in the same lane/VGPR slots as lrun), so only the running max
// needs cross-lane shuffles.
// ---------------------------------------------------------------------------
template <bool MASKED>
__device__ __forceinline__ void attn_step32(
    const int j0, const int m0, const int r, const int half,
    const _Float16* __restrict__ khead,   // kh + (b*T)*D + h*64
    const _Float16* __restrict__ vhead,   // vht + (b*H+h)*HD*T
    const v16h qa0, const v16h qa1, const v16h ones, _Float16* pl,
    v8f (&oA)[4], float (&mrun)[8], float (&lrun)[8]) {
  // ---- S = Q @ K^T (two 16x16 tiles over 32 keys) ----
  v8f s0 = {}, s1 = {};
  const _Float16* kp = khead + (size_t)(j0 + r) * D_ + half * 8;
  s0 = wmma_f16(qa0, ldfrag(kp), s0);
  s0 = wmma_f16(qa1, ldfrag(kp + 32), s0);
  const _Float16* kp2 = kp + (size_t)16 * D_;
  s1 = wmma_f16(qa0, ldfrag(kp2), s1);
  s1 = wmma_f16(qa1, ldfrag(kp2 + 32), s1);

  // ---- scale (+ causal mask) + online max / exp ----
  float alpha[8];
#pragma unroll
  for (int g = 0; g < 8; ++g) {
    float a0 = s0[g] * SCALE_F;
    float a1 = s1[g] * SCALE_F;
    if (MASKED) {
      const int row = m0 + half * 8 + g;   // C-layout row
      if (j0 + r > row) a0 = NEG_LARGE;    // col = j0 + r
      if (j0 + 16 + r > row) a1 = NEG_LARGE;
    }
    float mc = fmaxf(a0, a1);
#pragma unroll
    for (int msk = 1; msk < 16; msk <<= 1) mc = fmaxf(mc, __shfl_xor(mc, msk, 32));
    const float mn = fmaxf(mrun[g], mc);
    alpha[g] = __expf(mrun[g] - mn);
    mrun[g] = mn;
    const float p0 = __expf(a0 - mn);
    const float p1 = __expf(a1 - mn);
    const int row = half * 8 + g;
    pl[row * 48 + r] = (_Float16)p0;
    pl[row * 48 + 16 + r] = (_Float16)p1;
  }

  // rescale accumulators
#pragma unroll
  for (int t = 0; t < 4; ++t)
#pragma unroll
    for (int g = 0; g < 8; ++g) oA[t][g] *= alpha[g];

  asm volatile("s_wait_dscnt 0x0" ::: "memory");

  // ---- P (A-frag via LDS relayout) @ V, plus P @ ones = row sums ----
  const v16h pa = ldfrag(pl + r * 48 + half * 8);
  v8f lz = {};
  lz = wmma_f16(pa, ones, lz);  // lz[g] = sum_k P[row,k]
#pragma unroll
  for (int t = 0; t < 4; ++t) {
    const _Float16* vp = vhead + (size_t)(t * 16 + r) * T_ + j0 + half * 8;
    oA[t] = wmma_f16(pa, ldfrag(vp), oA[t]);
  }
#pragma unroll
  for (int g = 0; g < 8; ++g) lrun[g] = lrun[g] * alpha[g] + lz[g];
}

// ---------------------------------------------------------------------------
// Causal flash attention. Grid: (T/64, B*H). Block: 128 threads = 4 waves.
// Each wave owns 16 query rows; 32 keys per step (9 WMMAs / step).
// ---------------------------------------------------------------------------
__global__ __launch_bounds__(128) void flash_attn_kernel(
    const _Float16* __restrict__ qh, const _Float16* __restrict__ kh,
    const _Float16* __restrict__ vht, float* __restrict__ oatt) {
  __shared__ _Float16 pbuf[4][16 * 48];  // per-wave P tile, padded stride 48

  const int lane = threadIdx.x & 31;
  const int wave = threadIdx.x >> 5;
  const int r = lane & 15;
  const int half = lane >> 4;
  const int bh = blockIdx.y;
  const int b = bh >> 4;
  const int h = bh & 15;
  const int m0 = blockIdx.x * 64 + wave * 16;
  const size_t tokbase = (size_t)b * T_;

  // Q A-fragments for d-chunks [0,32) and [32,64)
  const _Float16* qp = qh + (tokbase + m0 + r) * (size_t)D_ + h * 64 + half * 8;
  const v16h qa0 = ldfrag(qp);
  const v16h qa1 = ldfrag(qp + 32);

  v16h ones;
#pragma unroll
  for (int i = 0; i < 16; ++i) ones[i] = (_Float16)1.0f;

  v8f oA[4] = {};
  float mrun[8], lrun[8];
#pragma unroll
  for (int g = 0; g < 8; ++g) { mrun[g] = NEG_LARGE; lrun[g] = 0.0f; }

  _Float16* pl = &pbuf[wave][0];
  const _Float16* khead = kh + tokbase * (size_t)D_ + h * 64;
  const _Float16* vhead = vht + ((size_t)b * H_ + h) * (size_t)HD_ * T_;

  // Fully-unmasked blocks (strictly below the diagonal), then one masked tail.
  const int nfull = m0 >> 5;
  for (int i = 0; i < nfull; ++i)
    attn_step32<false>(i * 32, m0, r, half, khead, vhead, qa0, qa1, ones, pl,
                       oA, mrun, lrun);
  attn_step32<true>(nfull * 32, m0, r, half, khead, vhead, qa0, qa1, ones, pl,
                    oA, mrun, lrun);

  // ---- epilogue: normalize by row sums, store f32 [b,h,t,hd] ----
#pragma unroll
  for (int g = 0; g < 8; ++g) {
    const float inv = 1.0f / lrun[g];
    const int row = m0 + half * 8 + g;
    const size_t obase = (((size_t)b * H_ + h) * T_ + row) * (size_t)HD_;
#pragma unroll
    for (int t = 0; t < 4; ++t) oatt[obase + t * 16 + r] = oA[t][g] * inv;
  }
}

// ---------------------------------------------------------------------------
// KNN memory attention + gate + bf16 rounding. One wave per (b,t,h);
// lane covers 2 of the 64 head dims. Output f16 "combined" [b,t,D].
// ---------------------------------------------------------------------------
__global__ __launch_bounds__(128) void mem_combine_kernel(
    const _Float16* __restrict__ qh, const float* __restrict__ oatt,
    const float* __restrict__ mem_bank, const int* __restrict__ knn_idx,
    const float* __restrict__ gate_bias, _Float16* __restrict__ ch) {
  const int wid = blockIdx.x * 4 + (threadIdx.x >> 5);
  const int lane = threadIdx.x & 31;
  const int h = wid % H_;
  const int t = (wid / H_) % T_;
  const int b = wid / (H_ * T_);
  const int d0 = lane * 2;

  const size_t qoff = (size_t)(b * T_ + t) * D_ + h * 64 + d0;
  const float q0 = (float)qh[qoff];
  const float q1 = (float)qh[qoff + 1];

  int idx[K_];
  float sc[K_];
#pragma unroll
  for (int k = 0; k < K_; ++k) {
    idx[k] = knn_idx[(size_t)(b * T_ + t) * K_ + k];
    const float* mk =
        mem_bank + (((size_t)b * M_ + idx[k]) * 2 + 0) * (size_t)D_ + h * 64 + d0;
    float dp = q0 * mk[0] + q1 * mk[1];
#pragma unroll
    for (int m = 16; m; m >>= 1) dp += __shfl_xor(dp, m, 32);
    sc[k] = dp * SCALE_F;
  }

  const float mx = fmaxf(sc[0], fmaxf(sc[1], sc[2]));
  float w[K_];
  float wsum = 0.0f;
#pragma unroll
  for (int k = 0; k < K_; ++k) { w[k] = __expf(sc[k] - mx); wsum += w[k]; }
  const float winv = 1.0f / wsum;

  float a0 = 0.0f, a1 = 0.0f;
#pragma unroll
  for (int k = 0; k < K_; ++k) {
    const float* mv =
        mem_bank + (((size_t)b * M_ + idx[k]) * 2 + 1) * (size_t)D_ + h * 64 + d0;
    a0 += w[k] * winv * mv[0];
    a1 += w[k] * winv * mv[1];
  }

  const float g = gate_bias[h];
  const size_t ooff = (((size_t)b * H_ + h) * T_ + t) * (size_t)HD_ + d0;
  float c0 = a0 * g + oatt[ooff] * (1.0f - g);
  float c1 = a1 * g + oatt[ooff + 1] * (1.0f - g);
  c0 = __bfloat162float(__float2bfloat16(c0));  // reference's bf16 round-trip
  c1 = __bfloat162float(__float2bfloat16(c1));
  ch[qoff] = (_Float16)c0;
  ch[qoff + 1] = (_Float16)c1;
}

// ---------------------------------------------------------------------------
// Host launcher
// ---------------------------------------------------------------------------
extern "C" void kernel_launch(void* const* d_in, const int* in_sizes, int n_in,
                              void* d_out, int out_size, void* d_ws,
                              size_t ws_size, hipStream_t stream) {
  (void)in_sizes; (void)n_in; (void)out_size; (void)ws_size;
  const float* x    = (const float*)d_in[0];
  const float* Wq   = (const float*)d_in[1];
  const float* bq   = (const float*)d_in[2];
  const float* Wk   = (const float*)d_in[3];
  const float* bk   = (const float*)d_in[4];
  const float* Wv   = (const float*)d_in[5];
  const float* bv   = (const float*)d_in[6];
  const float* Wo   = (const float*)d_in[7];
  const float* bo   = (const float*)d_in[8];
  const float* gate = (const float*)d_in[9];
  const float* mem  = (const float*)d_in[10];
  const int*   knn  = (const int*)d_in[11];
  float* out = (float*)d_out;

  const size_t BT = (size_t)B_ * T_;  // 4096
  size_t off = 0;
  auto take = [&](size_t bytes) -> void* {
    void* p = (char*)d_ws + off;
    off += (bytes + 255) & ~(size_t)255;
    return p;
  };
  _Float16* xh  = (_Float16*)take(BT * D_ * 2);
  _Float16* wqh = (_Float16*)take((size_t)D_ * D_ * 2);
  _Float16* wkh = (_Float16*)take((size_t)D_ * D_ * 2);
  _Float16* wvh = (_Float16*)take((size_t)D_ * D_ * 2);
  _Float16* woh = (_Float16*)take((size_t)D_ * D_ * 2);
  _Float16* qhb = (_Float16*)take(BT * D_ * 2);
  _Float16* khb = (_Float16*)take(BT * D_ * 2);
  _Float16* vhb = (_Float16*)take(BT * D_ * 2);
  _Float16* vht = (_Float16*)take(BT * D_ * 2);
  _Float16* chb = (_Float16*)take(BT * D_ * 2);
  float*    oat = (float*)take(BT * D_ * 4);

  // Stage 0: f32 -> f16
  f32_to_f16_kernel<<<8192, 256, 0, stream>>>(x, xh, BT * D_);
  f32_to_f16_kernel<<<4096, 256, 0, stream>>>(Wq, wqh, (size_t)D_ * D_);
  f32_to_f16_kernel<<<4096, 256, 0, stream>>>(Wk, wkh, (size_t)D_ * D_);
  f32_to_f16_kernel<<<4096, 256, 0, stream>>>(Wv, wvh, (size_t)D_ * D_);
  f32_to_f16_kernel<<<4096, 256, 0, stream>>>(Wo, woh, (size_t)D_ * D_);

  // Stage 1: QKV projections (WMMA, async-LDS-staged W tiles)
  dim3 ggrid(D_ / 64, (int)(BT / 64));
  gemm_nt_kernel<true><<<ggrid, 128, 0, stream>>>(xh, wqh, bq, nullptr, qhb,
                                                  (int)BT, D_, D_);
  gemm_nt_kernel<true><<<ggrid, 128, 0, stream>>>(xh, wkh, bk, nullptr, khb,
                                                  (int)BT, D_, D_);
  gemm_nt_kernel<true><<<ggrid, 128, 0, stream>>>(xh, wvh, bv, nullptr, vhb,
                                                  (int)BT, D_, D_);

  // Stage 2: normalize q,k; transpose v
  normalize_rows_kernel<<<(int)BT, 256, 0, stream>>>(qhb);
  normalize_rows_kernel<<<(int)BT, 256, 0, stream>>>(khb);
  transpose_v_kernel<<<(int)BT, 256, 0, stream>>>(vhb, vht);

  // Stage 3: causal flash attention (WMMA)
  flash_attn_kernel<<<dim3(T_ / 64, B_ * H_), 128, 0, stream>>>(qhb, khb, vht, oat);

  // Stage 4: KNN memory attention + gate + bf16 rounding
  mem_combine_kernel<<<(B_ * T_ * H_) / 4, 128, 0, stream>>>(qhb, oat, mem, knn,
                                                             gate, chb);

  // Stage 5: output projection (WMMA, f32 out)
  gemm_nt_kernel<false><<<ggrid, 128, 0, stream>>>(chb, woh, bo, out, nullptr,
                                                   (int)BT, D_, D_);
}
